// GRUModel_19344532701628
// MI455X (gfx1250) — matrix operand order repaired and checked by reading, not compiled, not run on hardware
//
#include <hip/hip_runtime.h>
#include <hip/hip_bf16.h>
#include <math.h>
#include <stdint.h>

// ---------------------------------------------------------------------------
// Problem constants (reference: T=512, B=256, F=256, H1=128, H2=64, OUT=5)
// ---------------------------------------------------------------------------
#define T_DIM 512
#define B_DIM 256
#define F_DIM 256
#define H1_DIM 128
#define H2_DIM 64
#define OUT_DIM 5

typedef _Float16 v16h __attribute__((ext_vector_type(16)));
typedef _Float16 v8h  __attribute__((ext_vector_type(8)));
typedef _Float16 v4h  __attribute__((ext_vector_type(4)));
typedef float    v8f  __attribute__((ext_vector_type(8)));
typedef float    v4f  __attribute__((ext_vector_type(4)));
typedef int      v4i  __attribute__((ext_vector_type(4)));

// ---- CDNA5 async Global->LDS path (guarded: falls back to direct loads) ----
#if defined(__has_builtin)
#if __has_builtin(__builtin_amdgcn_global_load_async_to_lds_b128)
#define HAVE_ASYNC_LDS 1
#endif
#endif
#ifndef HAVE_ASYNC_LDS
#define HAVE_ASYNC_LDS 0
#endif

#if HAVE_ASYNC_LDS
typedef __attribute__((address_space(1))) v4i* g_v4i_ptr;
typedef __attribute__((address_space(3))) v4i* l_v4i_ptr;

__device__ __forceinline__ void async_ld_b128(const void* g, void* l) {
    // GLOBAL_LOAD_ASYNC_TO_LDS_B128: LDS[l] = MEM[g], tracked by ASYNCcnt.
    // Pointer casts go through uintptr_t to enter the numbered address
    // spaces; for LDS the flat-address low 32 bits are the LDS byte offset.
    g_v4i_ptr gp = (g_v4i_ptr)(uintptr_t)g;
    l_v4i_ptr lp = (l_v4i_ptr)(uint32_t)(uintptr_t)l;
    __builtin_amdgcn_global_load_async_to_lds_b128(gp, lp, /*offset=*/0, /*cpol=*/0);
}
#if __has_builtin(__builtin_amdgcn_s_wait_asynccnt)
#define WAIT_ASYNC(n) __builtin_amdgcn_s_wait_asynccnt(n)
#else
#define WAIT_ASYNC(n) asm volatile("s_wait_asynccnt %0" ::"n"(n))
#endif
#endif

__device__ __forceinline__ v8f wmma_f16(v16h a, v16h b, v8f c) {
    // v_wmma_f32_16x16x32_f16 : D = A(16x32 f16) * B(32x16 f16) + C(16x16 f32)
    return __builtin_amdgcn_wmma_f32_16x16x32_f16(
        /*neg_a=*/false, a, /*neg_b=*/false, b,
        /*c_mod=*/(short)0, c, /*reuse_a=*/false, /*reuse_b=*/false);
}

// Build a v16h A-fragment from two contiguous 8-half segments.
__device__ __forceinline__ v16h make_afrag(const _Float16* p) {
    v8h lo = *(const v8h*)(p);        // K = kbase + kb .. +7
    v8h hi = *(const v8h*)(p + 16);   // K = kbase + 16 + kb .. +7
    return __builtin_shufflevector(lo, hi, 0,1,2,3,4,5,6,7,8,9,10,11,12,13,14,15);
}

// ---------------------------------------------------------------------------
// Kernel: f32 -> f16 conversion, 4 elements/thread (b128 in, b64 out)
// ---------------------------------------------------------------------------
__global__ void f32_to_f16_kernel(const v4f* __restrict__ src,
                                  v4h* __restrict__ dst, size_t n4) {
    size_t i = (size_t)blockIdx.x * blockDim.x + threadIdx.x;
    if (i >= n4) return;
    v4f v = src[i];
    v4h o;
    o.x = (_Float16)v.x; o.y = (_Float16)v.y;
    o.z = (_Float16)v.z; o.w = (_Float16)v.w;
    dst[i] = o;
}

// ---------------------------------------------------------------------------
// Kernel: pre-swizzle a weight W[N][K] (PyTorch w[g][k]) into WMMA B-fragment
// order (f16).  Fragment for (kt, nt):
//   frag[((kt*NT + nt)*32 + lane)*16 + e] = W[nt*16 + (lane&15)]
//                                            [kt*32 + (lane<16?0:8) + (e<8?e:8+e)]
// matching the 16-bit B 32x16 VGPR layout (lanes 0-15: K=0..7,16..23,
// lanes 16-31: K=8..15,24..31, 2 halves per VGPR).
// ---------------------------------------------------------------------------
__global__ void preswizzle_kernel(const float* __restrict__ W,
                                  _Float16* __restrict__ frag,
                                  int NT, int KT, int Kdim) {
    int idx = blockIdx.x * 256 + threadIdx.x;
    int total = KT * NT * 512;
    if (idx >= total) return;
    int e    = idx & 15;
    int lane = (idx >> 4) & 31;
    int nt   = (idx >> 9) % NT;
    int kt   = idx / (512 * NT);
    int n = nt * 16 + (lane & 15);
    int k = kt * 32 + ((lane < 16) ? 0 : 8) + ((e < 8) ? e : 8 + e);
    frag[idx] = (_Float16)W[n * Kdim + k];
}

// ---------------------------------------------------------------------------
// Kernel: C[M][N] = A[M][K](f16) * Bfrag(K x N, preswizzled f16) + bias[N]
// One wave computes one 16-row M tile x 4 consecutive 16-col N tiles.
// Block = 8 waves = 8 M tiles.  grid.x = M/128, grid.y = NT/4.
// ---------------------------------------------------------------------------
template <int K, int N>
__global__ void gemm_bias_kernel(const _Float16* __restrict__ A,
                                 const _Float16* __restrict__ fragB,
                                 const float* __restrict__ bias,
                                 float* __restrict__ C) {
    constexpr int KT = K / 32;
    constexpr int NT = N / 16;
    const int lane  = threadIdx.x & 31;
    const int wave  = threadIdx.x >> 5;
    const int mtile = blockIdx.x * 8 + wave;
    const int ng    = blockIdx.y * 4;       // first of 4 n-tiles for this wave
    const int mbase = mtile * 16;
    const int mrow  = mbase + (lane & 15);
    const int kb    = (lane < 16) ? 0 : 8;  // A-fragment sub-offset per lane half

    v8f acc0 = {}, acc1 = {}, acc2 = {}, acc3 = {};
    const _Float16* arow = A + (size_t)mrow * K + kb;

#pragma unroll
    for (int kt = 0; kt < KT; ++kt) {
        if (kt + 1 < KT) __builtin_prefetch(arow + (kt + 1) * 32, 0, 3);
        v16h a = make_afrag(arow + kt * 32);
        const _Float16* bp = fragB + ((size_t)(kt * NT + ng) * 32 + lane) * 16;
        v16h b0 = *(const v16h*)(bp);
        v16h b1 = *(const v16h*)(bp + 512);
        v16h b2 = *(const v16h*)(bp + 1024);
        v16h b3 = *(const v16h*)(bp + 1536);
        acc0 = wmma_f16(a, b0, acc0);
        acc1 = wmma_f16(a, b1, acc1);
        acc2 = wmma_f16(a, b2, acc2);
        acc3 = wmma_f16(a, b3, acc3);
    }

    const int col   = lane & 15;
    const int rbase = mbase + ((lane < 16) ? 0 : 8);
    float bb0 = bias[(ng + 0) * 16 + col];
    float bb1 = bias[(ng + 1) * 16 + col];
    float bb2 = bias[(ng + 2) * 16 + col];
    float bb3 = bias[(ng + 3) * 16 + col];
#pragma unroll
    for (int r = 0; r < 8; ++r) {
        float* crow = C + (size_t)(rbase + r) * N;
        crow[(ng + 0) * 16 + col] = acc0[r] + bb0;
        crow[(ng + 1) * 16 + col] = acc1[r] + bb1;
        crow[(ng + 2) * 16 + col] = acc2[r] + bb2;
        crow[(ng + 3) * 16 + col] = acc3[r] + bb3;
    }
}

// ---------------------------------------------------------------------------
// Kernel: persistent GRU recurrence over T timesteps for a 16-row batch tile.
// Per step:  HP = h * Whh^T + b_hh  (WMMA, M=16, N=3H, K=H, h staged in LDS)
//            gates from XP[t] (f32, precomputed) + HP  -> h_new
// XP[t] tiles (contiguous 16*3H floats per block) are double-buffered in LDS
// via CDNA5 async global->LDS loads, prefetched one timestep ahead so the
// global read latency is hidden behind the recurrent WMMA matmul.
// grid.x = B/16, blockDim = 256 (8 waves).
// ---------------------------------------------------------------------------
template <int H>
__global__ void gru_recurrence_kernel(const float* __restrict__ XP,      // [T][B][3H]
                                      const _Float16* __restrict__ fragU,// whh fragments
                                      const float* __restrict__ b_hh,    // [3H]
                                      _Float16* __restrict__ hseq,       // [T][B][H] or null
                                      float* __restrict__ hlast_seq,     // [T][H] or null
                                      int T, int Bsz) {
    constexpr int N  = 3 * H;
    constexpr int KT = H / 32;
    constexpr int NT = N / 16;
    constexpr int NLOADS = (16 * N * 4) / (256 * 16);  // b128 chunks per thread

    __shared__ _Float16 hbuf[16 * H];   // current hidden state tile (f16)
    __shared__ float    hpbuf[16 * N];  // h * Whh^T + b_hh
#if HAVE_ASYNC_LDS
    __shared__ float    xpbuf[2][16 * N];  // double-buffered XP[t] tile
#endif

    const int tid   = threadIdx.x;
    const int lane  = tid & 31;
    const int wave  = tid >> 5;
    const int rbase = blockIdx.x * 16;
    const int m16   = lane & 15;
    const int kb    = (lane < 16) ? 0 : 8;
    const int rr    = (lane < 16) ? 0 : 8;

    for (int i = tid; i < 16 * H; i += 256) hbuf[i] = (_Float16)0.f;
    __syncthreads();

#if HAVE_ASYNC_LDS
    {   // prefetch XP[0] into buffer 0
        const char* src = (const char*)(XP + ((size_t)0 * Bsz + rbase) * N);
        char* dst = (char*)xpbuf[0];
#pragma unroll
        for (int k = 0; k < NLOADS; ++k) {
            int o = (tid + k * 256) * 16;
            async_ld_b128(src + o, dst + o);
        }
    }
#endif

    for (int t = 0; t < T; ++t) {
#if HAVE_ASYNC_LDS
        {   // prefetch XP[t+1] into the other buffer (wraps harmlessly at end)
            int tn = (t + 1 < T) ? (t + 1) : 0;
            const char* src = (const char*)(XP + ((size_t)tn * Bsz + rbase) * N);
            char* dst = (char*)xpbuf[(t + 1) & 1];
#pragma unroll
            for (int k = 0; k < NLOADS; ++k) {
                int o = (tid + k * 256) * 16;
                async_ld_b128(src + o, dst + o);
            }
        }
#endif
        // ---- recurrent matmul: HP = h(16xH) * U(Hx3H) ----
        v16h afr[KT];
#pragma unroll
        for (int kt = 0; kt < KT; ++kt)
            afr[kt] = make_afrag(&hbuf[m16 * H + kt * 32 + kb]);

        for (int nt = wave; nt < NT; nt += 8) {   // wave-uniform -> EXEC all-1s
            v8f acc = {};
#pragma unroll
            for (int kt = 0; kt < KT; ++kt) {
                v16h b = *(const v16h*)(fragU + ((size_t)(kt * NT + nt) * 32 + lane) * 16);
                acc = wmma_f16(afr[kt], b, acc);
            }
            float bb = b_hh[nt * 16 + m16];
#pragma unroll
            for (int r = 0; r < 8; ++r)
                hpbuf[(rr + r) * N + nt * 16 + m16] = acc[r] + bb;
        }
#if HAVE_ASYNC_LDS
        // Async loads complete in order: <= NLOADS outstanding means the
        // batch issued for step t (one iteration ago) has landed in LDS.
        WAIT_ASYNC(NLOADS);
#endif
        __syncthreads();

        // ---- gates + state update ----
#if HAVE_ASYNC_LDS
        const float* xp_t = xpbuf[t & 1];
#else
        const float* xp_t = XP + ((size_t)t * Bsz + rbase) * N;
#endif
        for (int i = tid; i < 16 * H; i += 256) {
            int m = i / H, j = i % H;
            const float* xr = xp_t + (size_t)m * N;
            float xrv = xr[j], xzv = xr[H + j], xnv = xr[2 * H + j];
            float hr = hpbuf[m * N + j];
            float hz = hpbuf[m * N + H + j];
            float hn = hpbuf[m * N + 2 * H + j];
            float rg = 1.f / (1.f + expf(-(xrv + hr)));
            float zg = 1.f / (1.f + expf(-(xzv + hz)));
            float ng = tanhf(xnv + rg * hn);
            float hprev = (float)hbuf[m * H + j];
            float hnew  = (1.f - zg) * ng + zg * hprev;
            hbuf[m * H + j] = (_Float16)hnew;
            if (hseq)
                hseq[((size_t)t * Bsz + rbase + m) * H + j] = (_Float16)hnew;
            if (hlast_seq && (rbase + m == Bsz - 1))
                hlast_seq[(size_t)t * H + j] = hnew;
        }
        __syncthreads();
    }
}

// ---------------------------------------------------------------------------
// Kernel: out[t][o] = h2last[t] . lin_w[o] + lin_b[o]   (T=512, OUT=5, H2=64)
// ---------------------------------------------------------------------------
__global__ void final_linear_kernel(const float* __restrict__ h2last,
                                    const float* __restrict__ lin_w,
                                    const float* __restrict__ lin_b,
                                    float* __restrict__ out, int T) {
    int t = blockIdx.x * blockDim.x + threadIdx.x;
    if (t >= T) return;
    const float* h = h2last + (size_t)t * H2_DIM;
#pragma unroll
    for (int o = 0; o < OUT_DIM; ++o) {
        float s = lin_b[o];
        const float* w = lin_w + o * H2_DIM;
        for (int j = 0; j < H2_DIM; ++j) s += h[j] * w[j];
        out[t * OUT_DIM + o] = s;
    }
}

// ---------------------------------------------------------------------------
// Host driver
// ---------------------------------------------------------------------------
extern "C" void kernel_launch(void* const* d_in, const int* in_sizes, int n_in,
                              void* d_out, int out_size, void* d_ws, size_t ws_size,
                              hipStream_t stream) {
    const float* x     = (const float*)d_in[0];
    const float* w_ih1 = (const float*)d_in[1];
    const float* w_hh1 = (const float*)d_in[2];
    const float* b_ih1 = (const float*)d_in[3];
    const float* b_hh1 = (const float*)d_in[4];
    const float* w_ih2 = (const float*)d_in[5];
    const float* w_hh2 = (const float*)d_in[6];
    const float* b_ih2 = (const float*)d_in[7];
    const float* b_hh2 = (const float*)d_in[8];
    const float* lin_w = (const float*)d_in[9];
    const float* lin_b = (const float*)d_in[10];
    float* out = (float*)d_out;

    char* ws = (char*)d_ws;
    // Workspace layout (regions reused once their producer/consumer is done):
    //   [0 .. 64MB)        : x_f16  (K2 input)  -> later h1_f16 (K3 out / K4 in)
    //   [64MB .. 256MB)    : xp1 f32 (K2 out / K3 in) -> later xp2 f32 (K4/K5)
    //   [256MB .. )        : weight fragments + h2last
    const size_t M = (size_t)T_DIM * B_DIM;                     // 131072 rows
    size_t off = 0;
    _Float16* xh  = (_Float16*)(ws + off);                      // M*F f16
    _Float16* h1h = (_Float16*)(ws + off);                      // M*H1 f16 (alias)
    off += M * F_DIM * sizeof(_Float16);                        // 64 MB
    float* xp1 = (float*)(ws + off);                            // M*3H1 f32
    float* xp2 = (float*)(ws + off);                            // M*3H2 f32 (alias)
    off += M * 3 * H1_DIM * sizeof(float);                      // 192 MB
    _Float16* fWih1 = (_Float16*)(ws + off); off += (size_t)8 * 24 * 512 * 2;
    _Float16* fWhh1 = (_Float16*)(ws + off); off += (size_t)4 * 24 * 512 * 2;
    _Float16* fWih2 = (_Float16*)(ws + off); off += (size_t)4 * 12 * 512 * 2;
    _Float16* fWhh2 = (_Float16*)(ws + off); off += (size_t)2 * 12 * 512 * 2;
    float* h2last = (float*)(ws + off);      off += (size_t)T_DIM * H2_DIM * 4;

    // 1) x -> f16 (vectorized x4)
    {
        size_t n4 = (M * F_DIM) / 4;
        f32_to_f16_kernel<<<(unsigned)((n4 + 255) / 256), 256, 0, stream>>>(
            (const v4f*)x, (v4h*)xh, n4);
    }
    // 2) pre-swizzle all weights into WMMA B-fragment order
    preswizzle_kernel<<<(8 * 24 * 512) / 256, 256, 0, stream>>>(w_ih1, fWih1, 24, 8, F_DIM);
    preswizzle_kernel<<<(4 * 24 * 512) / 256, 256, 0, stream>>>(w_hh1, fWhh1, 24, 4, H1_DIM);
    preswizzle_kernel<<<(4 * 12 * 512) / 256, 256, 0, stream>>>(w_ih2, fWih2, 12, 4, H1_DIM);
    preswizzle_kernel<<<(2 * 12 * 512) / 256, 256, 0, stream>>>(w_hh2, fWhh2, 12, 2, H2_DIM);

    // 3) layer-1 input projection: xp1 = x * w_ih1^T + b_ih1   (WMMA)
    gemm_bias_kernel<F_DIM, 3 * H1_DIM>
        <<<dim3((unsigned)(M / 128), 6), 256, 0, stream>>>(xh, fWih1, b_ih1, xp1);

    // 4) layer-1 recurrence (persistent, 16 batch-tiles of 16 rows)
    gru_recurrence_kernel<H1_DIM>
        <<<B_DIM / 16, 256, 0, stream>>>(xp1, fWhh1, b_hh1, h1h, nullptr, T_DIM, B_DIM);

    // 5) layer-2 input projection: xp2 = h1 * w_ih2^T + b_ih2  (WMMA)
    gemm_bias_kernel<H1_DIM, 3 * H2_DIM>
        <<<dim3((unsigned)(M / 128), 3), 256, 0, stream>>>(h1h, fWih2, b_ih2, xp2);

    // 6) layer-2 recurrence; capture batch row (B-1) per timestep
    gru_recurrence_kernel<H2_DIM>
        <<<B_DIM / 16, 256, 0, stream>>>(xp2, fWhh2, b_hh2, nullptr, h2last, T_DIM, B_DIM);

    // 7) head: out = h2[:, -1, :] @ lin_w^T + lin_b
    final_linear_kernel<<<(T_DIM + 255) / 256, 256, 0, stream>>>(h2last, lin_w, lin_b, out, T_DIM);
}
